// RNN_89919435309085
// MI455X (gfx1250) — matrix operand order repaired and checked by reading, not compiled, not run on hardware
//
#include <hip/hip_runtime.h>
#include <cstdint>

#define Hdim 512
#define Edim 512
#define NB   32
#define TT   128
#define VV   50257
#define NCHUNK 197
#define VPAD (NCHUNK * 256)        // 50432
#define NROW 4096                  // 128 h-slots x 32 batch (slot 0 = initial h)
#define G3   (3 * Hdim)            // 1536

typedef __attribute__((ext_vector_type(16))) __bf16 v16bf;
typedef __attribute__((ext_vector_type(8)))  float  v8f;

// ---- WMMA helpers -----------------------------------------------------------

__device__ __forceinline__ v8f wmma_bf16(v16bf a, v16bf b, v8f c) {
    return __builtin_amdgcn_wmma_f32_16x16x32_bf16(
        false, a, false, b, (short)0, c, false, false);
}

// A-matrix 16x32 bf16 tile (MxK), row stride 512.
__device__ __forceinline__ v16bf loadA_bf(const __bf16* __restrict__ base, int lane, int k0) {
    int m = lane & 15, g = lane >> 4;
    const __bf16* p = base + m * 512 + k0 + g * 8;
    v16bf a;
#pragma unroll
    for (int i = 0; i < 8; ++i) { a[i] = p[i]; a[i + 8] = p[16 + i]; }
    return a;
}

// Same A layout from fp32 rows (per-lane gathered pointer), converting to bf16.
__device__ __forceinline__ v16bf loadA_cvt(const float* __restrict__ rowp, int g, int k0) {
    const float* p = rowp + k0 + g * 8;
    v16bf a;
#pragma unroll
    for (int i = 0; i < 8; ++i) { a[i] = (__bf16)p[i]; a[i + 8] = (__bf16)p[16 + i]; }
    return a;
}

// B-matrix 32x16 bf16 tile (KxN): B[k][n] = W[(n0+n)*512+k]; two 16B loads (k = g*8+i, 16+g*8+i).
__device__ __forceinline__ v16bf loadB_bf(const __bf16* __restrict__ base, int lane, int k0) {
    int n = lane & 15, g = lane >> 4;
    typedef __attribute__((ext_vector_type(8))) __bf16 v8bf;
    const __bf16* p = base + (long)n * 512 + k0 + g * 8;
    v8bf lo = *(const v8bf*)p, hi = *(const v8bf*)(p + 16);
    return __builtin_shufflevector(lo, hi, 0,1,2,3,4,5,6,7,8,9,10,11,12,13,14,15);
}

// ---- grid barrier (monotonic phase, 8 co-resident blocks) -------------------

__device__ __forceinline__ void grid_barrier(int* bar, int nblocks, int phase) {
    __syncthreads();
    if (threadIdx.x == 0) {
        __threadfence();
        int v = atomicAdd(&bar[0], 1);
        if (v == nblocks - 1) {
            bar[0] = 0;
            __threadfence();
            atomicExch(&bar[1], phase);
        } else {
            while (atomicAdd(&bar[1], 0) < phase) { __builtin_amdgcn_s_sleep(1); }
        }
        __threadfence();
    }
    __syncthreads();
}

// ---- pre-pass ---------------------------------------------------------------

__global__ void cvt_kernel(const float* __restrict__ src, __bf16* __restrict__ dst,
                           long nsrc, long ndst) {
    long i = (long)blockIdx.x * blockDim.x + threadIdx.x;
    if (i < ndst) dst[i] = (i < nsrc) ? (__bf16)src[i] : (__bf16)0.0f;
}

__global__ void init_kernel(const float* __restrict__ h0,
                            float* __restrict__ H32all, __bf16* __restrict__ HallBF,
                            int* __restrict__ bar) {
    int i = blockIdx.x * blockDim.x + threadIdx.x;
    if (i < NB * Hdim) { float v = h0[i]; H32all[i] = v; HallBF[i] = (__bf16)v; }
    if (i < 2) bar[i] = 0;
}

// ---- batched input-gate GEMM: gi[t*32+b] = emb(words[b,t]) @ W_ih^T + b_ih --
// grid (32, 16): 128-row blocks x 96-col groups; wave = 16 rows x 6 N-tiles,
// software-pipelined (next k-step's gathered A + B tiles in flight).

__global__ void gi_kernel(const int* __restrict__ words,
                          const float* __restrict__ embed_W,
                          const __bf16* __restrict__ Wih,
                          const float* __restrict__ b_ih,
                          float* __restrict__ gi_all) {
    int lane = threadIdx.x & 31, wv = threadIdx.x >> 5;
    int rb = blockIdx.x, cg = blockIdx.y;
    int m = lane & 15, g = lane >> 4;

    int gr = rb * 128 + wv * 16 + m;       // global row (t*32+b), t in 0..127
    int tt = gr >> 5, b = gr & 31;
    int widx = words[b * TT + tt];
    const float* xrow = embed_W + (long)widx * Edim;
    const __bf16* Bbase = Wih + (long)cg * 96 * Edim;

    v8f acc[6] = {};

    v16bf ac = loadA_cvt(xrow, g, 0);
    v16bf bc[6];
#pragma unroll
    for (int nt = 0; nt < 6; ++nt) bc[nt] = loadB_bf(Bbase + (long)nt * 16 * Edim, lane, 0);

#pragma unroll
    for (int k0 = 0; k0 < Edim; k0 += 32) {
        int k1 = (k0 + 32 < Edim) ? k0 + 32 : 0;   // clamped: last prefetch redundant
        v16bf an = loadA_cvt(xrow, g, k1);
        v16bf bn[6];
#pragma unroll
        for (int nt = 0; nt < 6; ++nt) bn[nt] = loadB_bf(Bbase + (long)nt * 16 * Edim, lane, k1);
#pragma unroll
        for (int nt = 0; nt < 6; ++nt) acc[nt] = wmma_bf16(ac, bc[nt], acc[nt]);
        ac = an;
#pragma unroll
        for (int nt = 0; nt < 6; ++nt) bc[nt] = bn[nt];
    }

    int n = lane & 15;
#pragma unroll
    for (int nt = 0; nt < 6; ++nt) {
        int col = cg * 96 + nt * 16 + n;
        float bias = b_ih[col];
#pragma unroll
        for (int v = 0; v < 8; ++v) {
            long row = (long)rb * 128 + wv * 16 + g * 8 + v;
            gi_all[row * G3 + col] = acc[nt][v] + bias;
        }
    }
}

// ---- persistent recurrence: 8 blocks x 256 thr = 64 waves, one h-tile each --
// inner loop software-pipelined: next k-step's A/B prefetched into registers

__global__ void rnn_kernel(const float* __restrict__ gi_all,
                           const __bf16* __restrict__ Whh,
                           const float* __restrict__ b_hh,
                           float* __restrict__ H32all,
                           __bf16* __restrict__ HallBF,
                           int* __restrict__ bar) {
    int lane = threadIdx.x & 31, wv = threadIdx.x >> 5;
    int id = blockIdx.x * 8 + wv;          // 0..63
    int mt = id & 1;                       // batch half
    int j  = (id >> 1) * 16;               // hidden-column tile
    int n = lane & 15, g = lane >> 4;

    const __bf16* Bhr = Whh + (long)(0 * Hdim + j) * Hdim;
    const __bf16* Bhz = Whh + (long)(1 * Hdim + j) * Hdim;
    const __bf16* Bhn = Whh + (long)(2 * Hdim + j) * Hdim;
    int col = j + n;
    float bhr = b_hh[col], bhz = b_hh[Hdim + col], bhn = b_hh[2 * Hdim + col];

    for (int t = 0; t < TT - 1; ++t) {
        const __bf16* Abase = HallBF + ((long)t * NB + mt * 16) * Hdim;
        v8f ahr = {}, ahz = {}, ahn = {};

        v16bf ac  = loadA_bf(Abase, lane, 0);
        v16bf brc = loadB_bf(Bhr, lane, 0);
        v16bf bzc = loadB_bf(Bhz, lane, 0);
        v16bf bnc = loadB_bf(Bhn, lane, 0);
#pragma unroll
        for (int k0 = 0; k0 < Hdim; k0 += 32) {
            int k1 = (k0 + 32 < Hdim) ? k0 + 32 : 0;   // clamped: last prefetch redundant
            v16bf an  = loadA_bf(Abase, lane, k1);
            v16bf brn = loadB_bf(Bhr, lane, k1);
            v16bf bzn = loadB_bf(Bhz, lane, k1);
            v16bf bnn = loadB_bf(Bhn, lane, k1);
            ahr = wmma_bf16(ac, brc, ahr);
            ahz = wmma_bf16(ac, bzc, ahz);
            ahn = wmma_bf16(ac, bnc, ahn);
            ac = an; brc = brn; bzc = bzn; bnc = bnn;
        }
#pragma unroll
        for (int v = 0; v < 8; ++v) {
            int rb = mt * 16 + g * 8 + v;                        // batch index
            long gir = ((long)t * NB + rb) * G3;
            float gr_ = gi_all[gir + col];
            float gz_ = gi_all[gir + Hdim + col];
            float gn_ = gi_all[gir + 2 * Hdim + col];
            float ho  = H32all[((long)t * NB + rb) * Hdim + col];
            float rg = 1.0f / (1.0f + expf(-(gr_ + ahr[v] + bhr)));
            float zg = 1.0f / (1.0f + expf(-(gz_ + ahz[v] + bhz)));
            float ng = tanhf(gn_ + rg * (ahn[v] + bhn));
            float hn = (1.0f - zg) * ng + zg * ho;
            long orow = ((long)(t + 1) * NB + rb) * Hdim + col;
            H32all[orow] = hn;
            HallBF[orow] = (__bf16)hn;
        }
        grid_barrier(bar, gridDim.x, t + 1);
    }
}

// ---- batched emit GEMM + chunk log-sum-exp ----------------------------------
// grid (197, 32): 256-col chunk x 128-row block; 512 thr = 16 waves,
// wave = 32 rows x 64 cols; double-buffered A/B register pipeline.

__global__ void emit_kernel(const __bf16* __restrict__ HallBF,
                            const __bf16* __restrict__ Wbf,
                            const float* __restrict__ emit_b,
                            float* __restrict__ chunk_m,
                            float* __restrict__ chunk_s) {
    __shared__ float redm[128 * 4];
    __shared__ float rowM[128];
    __shared__ float reds[128 * 4];

    int lane = threadIdx.x & 31, wv = threadIdx.x >> 5;
    int c0 = blockIdx.x * 256;
    int rb = blockIdx.y;
    int wm = wv & 3;        // row group (32 rows)
    int wn = wv >> 2;       // col group (64 cols)

    const __bf16* A0 = HallBF + ((long)rb * 128 + wm * 32) * Hdim;
    const __bf16* A1 = A0 + 16 * Hdim;
    const __bf16* Bb = Wbf + (long)(c0 + wn * 64) * Hdim;

    v8f acc0[4] = {}, acc1[4] = {};

    v16bf a0c = loadA_bf(A0, lane, 0);
    v16bf a1c = loadA_bf(A1, lane, 0);
    v16bf bc[4];
#pragma unroll
    for (int nt = 0; nt < 4; ++nt) bc[nt] = loadB_bf(Bb + (long)nt * 16 * Hdim, lane, 0);

#pragma unroll
    for (int k0 = 0; k0 < Hdim; k0 += 32) {
        int k1 = (k0 + 32 < Hdim) ? k0 + 32 : 0;       // clamped: last prefetch redundant
        if (k0 + 128 < Hdim) {                          // far stream-ahead hint
            __builtin_prefetch(Bb + (long)(lane & 15) * Hdim + k0 + 128, 0, 0);
        }
        v16bf a0n = loadA_bf(A0, lane, k1);
        v16bf a1n = loadA_bf(A1, lane, k1);
        v16bf bn[4];
#pragma unroll
        for (int nt = 0; nt < 4; ++nt) bn[nt] = loadB_bf(Bb + (long)nt * 16 * Hdim, lane, k1);
#pragma unroll
        for (int nt = 0; nt < 4; ++nt) {
            acc0[nt] = wmma_bf16(a0c, bc[nt], acc0[nt]);
            acc1[nt] = wmma_bf16(a1c, bc[nt], acc1[nt]);
        }
        a0c = a0n; a1c = a1n;
#pragma unroll
        for (int nt = 0; nt < 4; ++nt) bc[nt] = bn[nt];
    }

    int n = lane & 15, g = lane >> 4;
    float bias[4]; bool valid[4];
#pragma unroll
    for (int nt = 0; nt < 4; ++nt) {
        int gcol = c0 + wn * 64 + nt * 16 + n;
        valid[nt] = (gcol < VV);
        bias[nt]  = valid[nt] ? emit_b[gcol] : 0.0f;
    }

    // per-row max over this wave's 64 columns, then cross-wave via LDS
#pragma unroll
    for (int mt = 0; mt < 2; ++mt) {
        const v8f* ap = mt ? acc1 : acc0;
#pragma unroll
        for (int v = 0; v < 8; ++v) {
            float mx = -3.0e38f;
#pragma unroll
            for (int nt = 0; nt < 4; ++nt)
                if (valid[nt]) mx = fmaxf(mx, ap[nt][v] + bias[nt]);
#pragma unroll
            for (int off = 1; off < 16; off <<= 1)
                mx = fmaxf(mx, __shfl_xor(mx, off, 32));
            if (n == 0) redm[(wm * 32 + mt * 16 + g * 8 + v) * 4 + wn] = mx;
        }
    }
    __syncthreads();
    if (threadIdx.x < 128) {
        float M = redm[threadIdx.x * 4];
#pragma unroll
        for (int i = 1; i < 4; ++i) M = fmaxf(M, redm[threadIdx.x * 4 + i]);
        rowM[threadIdx.x] = M;
    }
    __syncthreads();

#pragma unroll
    for (int mt = 0; mt < 2; ++mt) {
        const v8f* ap = mt ? acc1 : acc0;
#pragma unroll
        for (int v = 0; v < 8; ++v) {
            int lr = wm * 32 + mt * 16 + g * 8 + v;
            float M = rowM[lr];
            float s = 0.0f;
#pragma unroll
            for (int nt = 0; nt < 4; ++nt)
                if (valid[nt]) s += expf(ap[nt][v] + bias[nt] - M);
#pragma unroll
            for (int off = 1; off < 16; off <<= 1)
                s += __shfl_xor(s, off, 32);
            if (n == 0) reds[lr * 4 + wn] = s;
        }
    }
    __syncthreads();
    if (threadIdx.x < 128) {
        int gr = rb * 128 + threadIdx.x;
        float S = reds[threadIdx.x * 4] + reds[threadIdx.x * 4 + 1]
                + reds[threadIdx.x * 4 + 2] + reds[threadIdx.x * 4 + 3];
        chunk_m[(long)blockIdx.x * NROW + gr] = rowM[threadIdx.x];
        chunk_s[(long)blockIdx.x * NROW + gr] = S;
    }
}

// ---- target logits: one wave per row (rows 32..4095) ------------------------

__global__ void tgt_kernel(const __bf16* __restrict__ HallBF,
                           const __bf16* __restrict__ Wbf,
                           const float* __restrict__ emit_b,
                           const int* __restrict__ words,
                           float* __restrict__ tgt) {
    int wid = (blockIdx.x * blockDim.x + threadIdx.x) >> 5;
    int lane = threadIdx.x & 31;
    int gr = 32 + wid;
    if (gr >= NROW) return;
    int tt = (gr >> 5) - 1, b = gr & 31;
    int wt = words[b * TT + tt + 1];
    const __bf16* hp = HallBF + (long)gr * Hdim;
    const __bf16* wp = Wbf + (long)wt * Hdim;
    float s = 0.0f;
#pragma unroll
    for (int i = 0; i < 16; ++i)
        s += (float)hp[lane * 16 + i] * (float)wp[lane * 16 + i];
#pragma unroll
    for (int off = 16; off >= 1; off >>= 1)
        s += __shfl_xor(s, off, 32);
    if (lane == 0) tgt[gr] = s + emit_b[wt];
}

// ---- merge chunk logsumexps -> per-(t,b) log-likelihood ---------------------

__global__ void merge_kernel(const float* __restrict__ chunk_m,
                             const float* __restrict__ chunk_s,
                             const float* __restrict__ tgt,
                             float* __restrict__ ll_tb) {
    int idx = blockIdx.x * blockDim.x + threadIdx.x;
    if (idx >= (TT - 1) * NB) return;
    int gr = 32 + idx;
    float M = -3.0e38f;
    for (int c = 0; c < NCHUNK; ++c) M = fmaxf(M, chunk_m[(long)c * NROW + gr]);
    float S = 0.0f;
    for (int c = 0; c < NCHUNK; ++c)
        S += chunk_s[(long)c * NROW + gr] * expf(chunk_m[(long)c * NROW + gr] - M);
    ll_tb[idx] = tgt[gr] - (M + logf(S));
}

// ---- output assembly --------------------------------------------------------

__global__ void out_kernel(const float* __restrict__ ll_tb,
                           const float* __restrict__ H32all,
                           float* __restrict__ out) {
    int i = blockIdx.x * blockDim.x + threadIdx.x;
    if (i < NB) {
        float a = 0.0f;
        for (int t = 0; t < TT - 1; ++t) a += ll_tb[t * NB + i];
        out[i] = a * (1.0f / (float)(TT - 1));
    }
    if (i < NB * Hdim) out[NB + i] = H32all[(long)(TT - 1) * NB * Hdim + i];
}

// ---- host launch ------------------------------------------------------------

extern "C" void kernel_launch(void* const* d_in, const int* in_sizes, int n_in,
                              void* d_out, int out_size, void* d_ws, size_t ws_size,
                              hipStream_t stream) {
    const int*   words   = (const int*)  d_in[0];
    const float* h0      = (const float*)d_in[1];
    const float* embed_W = (const float*)d_in[2];
    const float* W_ih    = (const float*)d_in[3];
    const float* W_hh    = (const float*)d_in[4];
    const float* b_ih    = (const float*)d_in[5];
    const float* b_hh    = (const float*)d_in[6];
    const float* emit_W  = (const float*)d_in[7];
    const float* emit_b  = (const float*)d_in[8];
    float* out = (float*)d_out;

    char* w = (char*)d_ws;
    size_t off = 0;
    auto alloc = [&](size_t bytes) -> void* {
        off = (off + 255) & ~(size_t)255;
        void* p = w + off;
        off += bytes;
        return p;
    };
    __bf16* emitWbf = (__bf16*)alloc((size_t)VPAD * Hdim * 2);      // 51.6 MB
    __bf16* Wihbf   = (__bf16*)alloc((size_t)G3 * Edim * 2);
    __bf16* Whhbf   = (__bf16*)alloc((size_t)G3 * Hdim * 2);
    __bf16* HallBF  = (__bf16*)alloc((size_t)NROW * Hdim * 2);      // 4.2 MB
    float*  H32all  = (float*) alloc((size_t)NROW * Hdim * 4);      // 8.4 MB
    float*  gi_all  = (float*) alloc((size_t)NROW * G3 * 4);        // 25.2 MB
    float*  chunk_m = (float*) alloc((size_t)NCHUNK * NROW * 4);    // 3.2 MB
    float*  chunk_s = (float*) alloc((size_t)NCHUNK * NROW * 4);
    float*  tgt     = (float*) alloc((size_t)NROW * 4);
    float*  ll_tb   = (float*) alloc((size_t)(TT - 1) * NB * 4);
    int*    bar     = (int*)   alloc(256);

    {   // bf16 conversions (emit_W zero-padded to VPAD rows)
        long nsrc = (long)VV * Hdim, ndst = (long)VPAD * Hdim;
        cvt_kernel<<<(int)((ndst + 255) / 256), 256, 0, stream>>>(emit_W, emitWbf, nsrc, ndst);
        long nw = (long)G3 * Edim;
        cvt_kernel<<<(int)((nw + 255) / 256), 256, 0, stream>>>(W_ih, Wihbf, nw, nw);
        cvt_kernel<<<(int)((nw + 255) / 256), 256, 0, stream>>>(W_hh, Whhbf, nw, nw);
    }
    init_kernel<<<(NB * Hdim + 255) / 256, 256, 0, stream>>>(h0, H32all, HallBF, bar);

    gi_kernel<<<dim3(32, 16), 256, 0, stream>>>(words, embed_W, Wihbf, b_ih, gi_all);
    rnn_kernel<<<8, 256, 0, stream>>>(gi_all, Whhbf, b_hh, H32all, HallBF, bar);
    emit_kernel<<<dim3(NCHUNK, 32), 512, 0, stream>>>(HallBF, emitWbf, emit_b,
                                                      chunk_m, chunk_s);
    tgt_kernel<<<((NROW - 32) * 32 + 255) / 256, 256, 0, stream>>>(HallBF, emitWbf,
                                                                   emit_b, words, tgt);
    merge_kernel<<<((TT - 1) * NB + 255) / 256, 256, 0, stream>>>(chunk_m, chunk_s,
                                                                  tgt, ll_tb);
    out_kernel<<<(NB * Hdim + 255) / 256, 256, 0, stream>>>(ll_tb, H32all, out);
}
